// LSTMClassifier_87024627352292
// MI455X (gfx1250) — matrix operand order repaired and checked by reading, not compiled, not run on hardware
//
#include <hip/hip_runtime.h>
#include <hip/hip_bf16.h>
#include <math.h>

typedef __attribute__((ext_vector_type(16))) _Float16 v16h;
typedef __attribute__((ext_vector_type(8)))  float    v8f;

#define T_DIM   256
#define B_DIM   64
#define IN_DIM  4096
#define H_DIM   110
#define G_DIM   440          // 4*H
#define NT      28           // ceil(440/16) N-tiles
#define KC1     128          // 4096/32 K-chunks for input GEMM
#define KC2     4            // ceil(110/32) -> pad K to 128 for recurrence
#define NCLS    20

#define XG_ELEMS  ((size_t)T_DIM * B_DIM * G_DIM)   // 7,208,960 f32
#define BHH_HALFS (KC2 * NT * 32 * 16)              // 57,344 f16 (packed B operand)

// Convert 8 contiguous f32 (two float4) into halves a[base..base+7]
__device__ __forceinline__ void cvt8(v16h& a, int base, const float4& u, const float4& v) {
    a[base + 0] = (_Float16)u.x; a[base + 1] = (_Float16)u.y;
    a[base + 2] = (_Float16)u.z; a[base + 3] = (_Float16)u.w;
    a[base + 4] = (_Float16)v.x; a[base + 5] = (_Float16)v.y;
    a[base + 6] = (_Float16)v.z; a[base + 7] = (_Float16)v.w;
}

// ---------------------------------------------------------------------------
// Kernel 1: xg[t*B+b][g] = x[t,b,:] . W_ih[g,:] + b_ih[g] + b_hh[g]
// Grid: 256 WGs x 256 threads. Each WG: 4 M-tiles (64 rows) x full N (28 tiles).
// Wave w owns N-tiles {w, w+8, w+16, w+24}. f16 WMMA, f32 accumulate.
// ---------------------------------------------------------------------------
__global__ __launch_bounds__(256) void k_input_gemm(
    const float* __restrict__ x, const float* __restrict__ W_ih,
    const float* __restrict__ b_ih, const float* __restrict__ b_hh,
    float* __restrict__ xg)
{
    const int tid   = threadIdx.x;
    const int lane  = tid & 31;
    const int w     = tid >> 5;        // wave 0..7
    const int lhalf = lane >> 4;       // 0: lanes 0-15, 1: lanes 16-31
    const int lmod  = lane & 15;
    const int mtBase = blockIdx.x * 4; // 4 M-tiles per WG

    v8f acc[4][4];                     // [ni][mi]
#pragma unroll
    for (int ni = 0; ni < 4; ++ni)
#pragma unroll
        for (int mi = 0; mi < 4; ++mi)
#pragma unroll
            for (int r = 0; r < 8; ++r) acc[ni][mi][r] = 0.0f;

    for (int kc = 0; kc < KC1; ++kc) {
        // ---- A operand: 4 M-tiles, contiguous k = 16*lhalf + s order (same as B) ----
        v16h a[4];
#pragma unroll
        for (int mi = 0; mi < 4; ++mi) {
            const float* rp = x + (size_t)((mtBase + mi) * 16 + lmod) * IN_DIM
                                + kc * 32 + lhalf * 16;
            float4 u0 = ((const float4*)rp)[0];
            float4 u1 = ((const float4*)rp)[1];
            float4 u2 = ((const float4*)(rp + 8))[0];
            float4 u3 = ((const float4*)(rp + 8))[1];
            cvt8(a[mi], 0, u0, u1);
            cvt8(a[mi], 8, u2, u3);
        }
        // ---- B operand per owned N-tile, then 4 WMMAs reusing it ----
#pragma unroll
        for (int ni = 0; ni < 4; ++ni) {
            const int nt = w + ni * 8;
            if (nt < NT) {                         // wave-uniform
                const int n = nt * 16 + lmod;
                v16h bm;
                if (n < G_DIM) {
                    const float* bp = W_ih + (size_t)n * IN_DIM + kc * 32 + lhalf * 16;
                    float4 c0 = ((const float4*)bp)[0];
                    float4 c1 = ((const float4*)bp)[1];
                    float4 c2 = ((const float4*)bp)[2];
                    float4 c3 = ((const float4*)bp)[3];
                    cvt8(bm, 0, c0, c1);
                    cvt8(bm, 8, c2, c3);
                } else {
#pragma unroll
                    for (int i = 0; i < 16; ++i) bm[i] = (_Float16)0.0f;
                }
#pragma unroll
                for (int mi = 0; mi < 4; ++mi)
                    acc[ni][mi] = __builtin_amdgcn_wmma_f32_16x16x32_f16(
                        false, a[mi], false, bm, (short)0, acc[ni][mi], false, false);
            }
        }
    }

    // ---- epilogue: add bias, store (C/D layout: lane=N+16*(M>=8), vgpr=M%8) ----
#pragma unroll
    for (int ni = 0; ni < 4; ++ni) {
        const int nt = w + ni * 8;
        if (nt < NT) {
            const int col = nt * 16 + lmod;
            float bias = 0.0f;
            if (col < G_DIM) bias = b_ih[col] + b_hh[col];
#pragma unroll
            for (int mi = 0; mi < 4; ++mi) {
#pragma unroll
                for (int r = 0; r < 8; ++r) {
                    const int m   = r + 8 * lhalf;
                    const int row = (mtBase + mi) * 16 + m;
                    if (col < G_DIM)
                        xg[(size_t)row * G_DIM + col] = acc[ni][mi][r] + bias;
                }
            }
        }
    }
}

// ---------------------------------------------------------------------------
// Kernel 1b: pack W_hh[440,110] f32 -> f16 B-operand lane layout, K padded to 128.
// Bhh index = ((kc*28 + nt)*32 + lane)*16 + hk ; lane<16 -> K=kc*32+hk, else +16.
// ---------------------------------------------------------------------------
__global__ __launch_bounds__(256) void k_pack_whh(
    const float* __restrict__ W_hh, _Float16* __restrict__ Bhh)
{
    const int idx = blockIdx.x * 256 + threadIdx.x;
    if (idx >= BHH_HALFS) return;
    const int hk   = idx & 15;
    const int lane = (idx >> 4) & 31;
    const int nt   = (idx >> 9) % NT;
    const int kc   = (idx >> 9) / NT;
    const int n = nt * 16 + (lane & 15);
    const int k = kc * 32 + ((lane >> 4) << 4) + hk;
    _Float16 v = (_Float16)0.0f;
    if (n < G_DIM && k < H_DIM) v = (_Float16)W_hh[n * H_DIM + k];
    Bhh[idx] = v;
}

// ---------------------------------------------------------------------------
// Kernel 2: sequential LSTM scan. 4 WGs, each owns 16 batch rows (one M-tile).
// Per step: gates = xg[t] + h @ W_hh^T via WMMA (A=h from LDS, B=packed Bhh),
// gates -> LDS -> fused cell update; capture h at t == lens[b]-1.
// ---------------------------------------------------------------------------
__global__ __launch_bounds__(256) void k_lstm_scan(
    const float* __restrict__ xg, const _Float16* __restrict__ Bhh,
    const int* __restrict__ lens, float* __restrict__ last_h)
{
    __shared__ float    gatesS[16][448];   // 28.0 KB (padded to 448 cols)
    __shared__ _Float16 hA[16][128];       //  4.0 KB, A-operand source (K padded)
    __shared__ float    cS[16][H_DIM];     //  6.9 KB

    const int tid   = threadIdx.x;
    const int lane  = tid & 31;
    const int w     = tid >> 5;
    const int lhalf = lane >> 4;
    const int lmod  = lane & 15;
    const int btile = blockIdx.x;          // 0..3

    for (int i = tid; i < 16 * 128; i += 256) hA[i >> 7][i & 127] = (_Float16)0.0f;
    for (int i = tid; i < 16 * H_DIM; i += 256) cS[i / H_DIM][i % H_DIM] = 0.0f;
    __syncthreads();

    for (int t = 0; t < T_DIM; ++t) {
        // ---- gate GEMM tiles ----
#pragma unroll
        for (int ni = 0; ni < 4; ++ni) {
            const int nt = w + ni * 8;
            if (nt < NT) {                          // wave-uniform
                const int col = nt * 16 + lmod;
                v8f acc;
#pragma unroll
                for (int r = 0; r < 8; ++r) {       // init with xg (bias included)
                    const int m = r + 8 * lhalf;
                    const int b = btile * 16 + m;
                    float v0 = 0.0f;
                    if (col < G_DIM)
                        v0 = xg[((size_t)t * B_DIM + b) * G_DIM + col];
                    acc[r] = v0;
                }
#pragma unroll
                for (int kc = 0; kc < KC2; ++kc) {
                    v16h a;                          // A = h tile from LDS
                    const _Float16* ap = &hA[lmod][kc * 32 + lhalf * 16];
#pragma unroll
                    for (int i = 0; i < 8; ++i) { a[i] = ap[i]; a[8 + i] = ap[8 + i]; }
                    v16h bm;                         // B = packed W_hh, coalesced 32B/lane
                    const _Float16* bp = Bhh + (((size_t)(kc * NT + nt) * 32 + lane) * 16);
#pragma unroll
                    for (int i = 0; i < 16; ++i) bm[i] = bp[i];
                    acc = __builtin_amdgcn_wmma_f32_16x16x32_f16(
                        false, a, false, bm, (short)0, acc, false, false);
                }
#pragma unroll
                for (int r = 0; r < 8; ++r) {
                    const int m = r + 8 * lhalf;
                    gatesS[m][nt * 16 + lmod] = acc[r];
                }
            }
        }
        __syncthreads();

        // ---- fused cell update: i,f,g,o -> c,h ----
        for (int idx = tid; idx < 16 * H_DIM; idx += 256) {
            const int m = idx / H_DIM, j = idx % H_DIM;
            const float ig = gatesS[m][j];
            const float fg = gatesS[m][H_DIM + j];
            const float gg = gatesS[m][2 * H_DIM + j];
            const float og = gatesS[m][3 * H_DIM + j];
            const float is = 1.0f / (1.0f + __expf(-ig));
            const float fs = 1.0f / (1.0f + __expf(-fg));
            const float gt = tanhf(gg);
            const float os = 1.0f / (1.0f + __expf(-og));
            const float c  = fs * cS[m][j] + is * gt;
            cS[m][j] = c;
            const float h = os * tanhf(c);
            hA[m][j] = (_Float16)h;
            const int b = btile * 16 + m;
            if (t == lens[b] - 1) last_h[b * H_DIM + j] = h;
        }
        __syncthreads();
    }
}

// ---------------------------------------------------------------------------
// Kernel 3: logits = last_h @ W_cls^T + b_cls, then log_softmax. 1 thread/row.
// ---------------------------------------------------------------------------
__global__ __launch_bounds__(64) void k_classifier(
    const float* __restrict__ last_h, const float* __restrict__ W_cls,
    const float* __restrict__ b_cls, float* __restrict__ out)
{
    const int b = threadIdx.x;            // 0..63
    float logits[NCLS];
    for (int c = 0; c < NCLS; ++c) {
        float s = b_cls[c];
        for (int j = 0; j < H_DIM; ++j)
            s += last_h[b * H_DIM + j] * W_cls[c * H_DIM + j];
        logits[c] = s;
    }
    float mx = logits[0];
    for (int c = 1; c < NCLS; ++c) mx = fmaxf(mx, logits[c]);
    float sum = 0.0f;
    for (int c = 0; c < NCLS; ++c) sum += __expf(logits[c] - mx);
    const float lse = mx + __logf(sum);
    for (int c = 0; c < NCLS; ++c) out[b * NCLS + c] = logits[c] - lse;
}

extern "C" void kernel_launch(void* const* d_in, const int* in_sizes, int n_in,
                              void* d_out, int out_size, void* d_ws, size_t ws_size,
                              hipStream_t stream)
{
    const float* x    = (const float*)d_in[0];
    const int*   lens = (const int*)  d_in[1];
    const float* Wih  = (const float*)d_in[2];
    const float* Whh  = (const float*)d_in[3];
    const float* bih  = (const float*)d_in[4];
    const float* bhh  = (const float*)d_in[5];
    const float* Wcls = (const float*)d_in[6];
    const float* bcls = (const float*)d_in[7];
    float* out = (float*)d_out;

    char* ws = (char*)d_ws;
    float*    xg     = (float*)ws;                                   // 28,835,840 B
    _Float16* BhhP   = (_Float16*)(ws + XG_ELEMS * 4);               //    114,688 B
    float*    last_h = (float*)(ws + XG_ELEMS * 4 + BHH_HALFS * 2);  //     28,160 B

    k_input_gemm<<<256, 256, 0, stream>>>(x, Wih, bih, bhh, xg);
    k_pack_whh<<<(BHH_HALFS + 255) / 256, 256, 0, stream>>>(Whh, BhhP);
    k_lstm_scan<<<4, 256, 0, stream>>>(xg, BhhP, lens, last_h);
    k_classifier<<<1, 64, 0, stream>>>(last_h, Wcls, bcls, out);
}
